// TemporalAxialAttention_463856468702
// MI455X (gfx1250) — matrix operand order repaired and checked by reading, not compiled, not run on hardware
//
#include <hip/hip_runtime.h>
#include <hip/hip_bf16.h>

// ---------------- types ----------------
typedef __attribute__((ext_vector_type(16))) _Float16 v16h;
typedef __attribute__((ext_vector_type(8)))  _Float16 v8h;
typedef __attribute__((ext_vector_type(8)))  float    v8f;

union H16 { v16h v; v8h h[2]; };

// ---------------- problem constants ----------------
#define DIMX     1024
#define E3       3072
#define HEADS    16
#define HD       64
#define TT       32
#define NTOK     36864      // B*T*H*W = 2*32*18*32
#define NSEQ     1152       // B*H*W   = 2*18*32
#define REMHW    576        // H*W     = 18*32
#define KSTEP    32

// ---------------- WMMA wrapper ----------------
__device__ __forceinline__ v8f wmma16(v16h a, v16h b, v8f c) {
    // D = A(16x32 f16) * B(32x16 f16) + C(16x16 f32)
    return __builtin_amdgcn_wmma_f32_16x16x32_f16(
        /*neg_a=*/false, a, /*neg_b=*/false, b,
        /*c_mod=*/(short)0, c, /*reuse_a=*/false, /*reuse_b=*/false);
}

// ---------------- async global -> LDS (CDNA5, ASYNCcnt-tracked) ----------
typedef __attribute__((address_space(3))) void lds_void;

__device__ __forceinline__ void async_ld_b128(void* lds_dst, const void* gsrc) {
    lds_void* l = (lds_void*)lds_dst;             // addrspacecast -> 32-bit LDS addr
    asm volatile("global_load_async_to_lds_b128 %0, %1, off"
                 : : "v"(l), "v"(gsrc) : "memory");
}

// ---------------- fragment loaders ----------------
// A fragment (16x32, f16) from row-major [M][K] f16 source.
// per-lane K chunks: {k0+half*8 .. +7, k0+16+half*8 .. +7}
__device__ __forceinline__ v16h ld_a16(const _Float16* row, int k0, int half) {
    H16 r;
    r.h[0] = *(const v8h*)(row + k0 + half * 8);
    r.h[1] = *(const v8h*)(row + k0 + 16 + half * 8);
    return r.v;
}

// B fragment (32x16, f16) from row-major [N][K] f16 source (B^T rows).
// per-lane K chunks: {k0+half*8 .. +7, k0+16+half*8 .. +7}
__device__ __forceinline__ v16h ld_b16(const _Float16* row, int k0, int half) {
    H16 r;
    r.h[0] = *(const v8h*)(row + k0 + half * 8);
    r.h[1] = *(const v8h*)(row + k0 + 16 + half * 8);
    return r.v;
}

// A fragment with fused f32 -> f16 conversion (row stride = KSTEP floats).
__device__ __forceinline__ v16h ld_a_f32cvt(const float* row, int half) {
    const float* p0 = row + half * 8;
    const float* p1 = row + 16 + half * 8;
    float4 a0 = *(const float4*)(p0);
    float4 a1 = *(const float4*)(p0 + 4);
    float4 b0 = *(const float4*)(p1);
    float4 b1 = *(const float4*)(p1 + 4);
    v16h v;
    v[0]=(_Float16)a0.x;  v[1]=(_Float16)a0.y;  v[2]=(_Float16)a0.z;  v[3]=(_Float16)a0.w;
    v[4]=(_Float16)a1.x;  v[5]=(_Float16)a1.y;  v[6]=(_Float16)a1.z;  v[7]=(_Float16)a1.w;
    v[8]=(_Float16)b0.x;  v[9]=(_Float16)b0.y;  v[10]=(_Float16)b0.z; v[11]=(_Float16)b0.w;
    v[12]=(_Float16)b1.x; v[13]=(_Float16)b1.y; v[14]=(_Float16)b1.z; v[15]=(_Float16)b1.w;
    return v;
}

// ---------------- kernel 0: f32 -> f16 convert ----------------
__global__ void cvt_f16_kernel(const float* __restrict__ in, _Float16* __restrict__ out, int n) {
    int i = blockIdx.x * blockDim.x + threadIdx.x;
    if (i < n) out[i] = (_Float16)in[i];
}

// ---------------- kernel 1: qkv = x @ w_qkv^T ----------------------------
// Block (256 thr = 8 waves) computes 128(M tokens) x 64(E feats).
// Per K-step(32): async-stage A(f32 128x32, 16KB) + B(f16 64x32, 4KB) into
// double-buffered LDS; 5 async b128 loads/thread; waves do 4 WMMAs each.
__global__ __launch_bounds__(256) void qkv_gemm_kernel(
        const float* __restrict__ x, const _Float16* __restrict__ wq,
        _Float16* __restrict__ qkv) {
    __shared__ __align__(16) float    As[2][128 * KSTEP];   // 2 x 16 KB
    __shared__ __align__(16) _Float16 Bs[2][64 * KSTEP];    // 2 x  4 KB

    const int tid  = threadIdx.x;
    const int wv   = tid >> 5;
    const int lane = tid & 31;
    const int half = lane >> 4;
    const int mn   = lane & 15;

    const int mblk  = blockIdx.x / 48;    // 288 super-tiles of 128 tokens
    const int ngrp  = blockIdx.x % 48;    // 48 groups of 64 features
    const int mbase = mblk * 128;
    const int ebase = ngrp * 64;

    auto issue = [&](int k0, int buf) {
        // A: 1024 chunks of 16B (128 rows x 8 chunks); 4 per thread
#pragma unroll
        for (int q = 0; q < 4; ++q) {
            int c   = tid + q * 256;
            int row = c >> 3;
            int col = (c & 7) * 4;                      // floats
            async_ld_b128(&As[buf][row * KSTEP + col],
                          x + (size_t)(mbase + row) * DIMX + k0 + col);
        }
        // B: 256 chunks of 16B (64 rows x 4 chunks); 1 per thread
        {
            int row = tid >> 2;
            int col = (tid & 3) * 8;                    // halves
            async_ld_b128(&Bs[buf][row * KSTEP + col],
                          wq + (size_t)(ebase + row) * DIMX + k0 + col);
        }
    };

    v8f acc[4] = {};
    int buf = 0;
    issue(0, 0);
    for (int k0 = 0; k0 < DIMX; k0 += KSTEP) {
        if (k0 + KSTEP < DIMX) {
            issue(k0 + KSTEP, buf ^ 1);                       // prefetch next
            asm volatile("s_wait_asynccnt 0x5" ::: "memory"); // prev batch (5) done
        } else {
            asm volatile("s_wait_asynccnt 0x0" ::: "memory");
        }
        __syncthreads();
        const float* arow = &As[buf][(wv * 16 + mn) * KSTEP];
        v16h a = ld_a_f32cvt(arow, half);
#pragma unroll
        for (int j = 0; j < 4; ++j) {
            v16h b = ld_b16(&Bs[buf][(j * 16 + mn) * KSTEP], 0, half);
            acc[j] = wmma16(a, b, acc[j]);
        }
        __syncthreads();
        buf ^= 1;
    }
    // D layout: reg r, lanes 0-15 -> M=r ; lanes 16-31 -> M=8+r ; N = mn
#pragma unroll
    for (int j = 0; j < 4; ++j)
#pragma unroll
        for (int r = 0; r < 8; ++r) {
            int tok = mbase + wv * 16 + r + half * 8;
            int e   = ebase + j * 16 + mn;
            qkv[(size_t)tok * E3 + e] = (_Float16)acc[j][r];
        }
}

// ---------------- kernel 2: attention (per seq, per head) ----------------
__global__ __launch_bounds__(128) void attn_kernel(
        const _Float16* __restrict__ qkv, _Float16* __restrict__ o) {
    __shared__ __align__(16) _Float16 sQ[4][TT * HD];
    __shared__ __align__(16) _Float16 sK[4][TT * HD];
    __shared__ __align__(16) _Float16 sV[4][TT * HD];
    __shared__ __align__(16) _Float16 sP[4][TT * TT];

    int wv   = threadIdx.x >> 5;
    int lane = threadIdx.x & 31;
    int gw   = blockIdx.x * 4 + wv;
    int n    = gw >> 4;          // sequence index (b*H + h)*W + w
    int head = gw & 15;
    int b    = n / REMHW;
    int rem  = n % REMHW;

    _Float16* Q = sQ[wv];
    _Float16* K = sK[wv];
    _Float16* V = sV[wv];
    _Float16* P = sP[wv];

    // ---- stage Q,K,V rows into LDS (lane t loads row t), fused RoPE ----
    {
        int t = lane;
        size_t tok = (size_t)(b * TT + t) * REMHW + rem;
        const _Float16* base = qkv + tok * E3 + head * HD;
        const _Float16* qp = base;
        const _Float16* kp = base + 1024;
        const _Float16* vp = base + 2048;
        const float scale = 0.125f;  // HEAD_DIM^-0.5
#pragma unroll
        for (int i = 0; i < 16; ++i) {           // rotary half: d < 32, pairs
            float fr  = __powf(10000.0f, -(float)i * (1.0f / 16.0f));
            float ang = (float)t * fr;
            float c = __cosf(ang), s = __sinf(ang);
            float q0 = (float)qp[2 * i], q1 = (float)qp[2 * i + 1];
            float k0 = (float)kp[2 * i], k1 = (float)kp[2 * i + 1];
            Q[t * HD + 2 * i]     = (_Float16)((q0 * c - q1 * s) * scale);
            Q[t * HD + 2 * i + 1] = (_Float16)((q1 * c + q0 * s) * scale);
            K[t * HD + 2 * i]     = (_Float16)(k0 * c - k1 * s);
            K[t * HD + 2 * i + 1] = (_Float16)(k1 * c + k0 * s);
        }
#pragma unroll
        for (int d = 32; d < 64; ++d) {          // pass-through half
            Q[t * HD + d] = (_Float16)((float)qp[d] * scale);
            K[t * HD + d] = kp[d];
        }
#pragma unroll
        for (int d = 0; d < 64; ++d) V[t * HD + d] = vp[d];
    }

    int half = lane >> 4;
    int mn   = lane & 15;

    // ---- S = Q * K^T : 2x2 tiles of 16x16, K-dim = 64 in 2 steps ----
    v8f c[2][2] = {};
#pragma unroll
    for (int kk = 0; kk < HD; kk += 32) {
        v16h aq[2], bk[2];
#pragma unroll
        for (int i = 0; i < 2; ++i) aq[i] = ld_a16(Q + (i * 16 + mn) * HD, kk, half);
#pragma unroll
        for (int j = 0; j < 2; ++j) bk[j] = ld_b16(K + (j * 16 + mn) * HD, kk, half);
#pragma unroll
        for (int i = 0; i < 2; ++i)
#pragma unroll
            for (int j = 0; j < 2; ++j) c[i][j] = wmma16(aq[i], bk[j], c[i][j]);
    }

    // ---- causal softmax over k (N axis) ----
#pragma unroll
    for (int i = 0; i < 2; ++i) {
#pragma unroll
        for (int r = 0; r < 8; ++r) {
            int m  = i * 16 + r + half * 8;     // query index
            int k0 = mn, k1 = 16 + mn;          // key indices in tiles 0,1
            float s0 = c[i][0][r], s1 = c[i][1][r];
            if (k0 > m) s0 = -1e30f;
            if (k1 > m) s1 = -1e30f;
            float mx = fmaxf(s0, s1);
#pragma unroll
            for (int off = 8; off > 0; off >>= 1) mx = fmaxf(mx, __shfl_xor(mx, off, 32));
            float e0 = __expf(s0 - mx), e1 = __expf(s1 - mx);
            float sm = e0 + e1;
#pragma unroll
            for (int off = 8; off > 0; off >>= 1) sm += __shfl_xor(sm, off, 32);
            float inv = __frcp_rn(sm);
            P[m * TT + k0] = (_Float16)(e0 * inv);
            P[m * TT + k1] = (_Float16)(e1 * inv);
        }
    }

    // ---- O = P(32x32) * V(32x64): 2 M-tiles x 4 N-tiles, single K step ----
    v16h ap[2];
#pragma unroll
    for (int i = 0; i < 2; ++i) ap[i] = ld_a16(P + (i * 16 + mn) * TT, 0, half);
    v8f co[2][4] = {};
#pragma unroll
    for (int jd = 0; jd < 4; ++jd) {
        int dcol  = jd * 16 + mn;               // N = feature column of V
        int tbase = half * 8;                   // per-lane K (=t) chunks: tbase..+7, tbase+16..+7
        v16h bv;
#pragma unroll
        for (int kk = 0; kk < 16; ++kk) bv[kk] = V[(tbase + (kk & 7) + (kk >> 3) * 16) * HD + dcol];
#pragma unroll
        for (int i = 0; i < 2; ++i) co[i][jd] = wmma16(ap[i], bv, co[i][jd]);
    }

    // ---- scatter O back to token-major f16 buffer [NTOK][1024] ----
#pragma unroll
    for (int i = 0; i < 2; ++i)
#pragma unroll
        for (int jd = 0; jd < 4; ++jd)
#pragma unroll
            for (int r = 0; r < 8; ++r) {
                int tq = i * 16 + r + half * 8;
                size_t tok = (size_t)(b * TT + tq) * REMHW + rem;
                o[tok * DIMX + head * HD + jd * 16 + mn] = (_Float16)co[i][jd][r];
            }
}

// ---------------- kernel 3: out = o @ w_out^T + b_out (fp32 out) ---------
// Block computes 128(M) x 64(N); async double-buffered LDS staging,
// 3 async b128 loads per thread per K-step.
__global__ __launch_bounds__(256) void out_gemm_kernel(
        const _Float16* __restrict__ o, const _Float16* __restrict__ wo,
        const float* __restrict__ bias, float* __restrict__ out) {
    __shared__ __align__(16) _Float16 As[2][128 * KSTEP];   // 2 x 8 KB
    __shared__ __align__(16) _Float16 Bs[2][64 * KSTEP];    // 2 x 4 KB

    const int tid  = threadIdx.x;
    const int wv   = tid >> 5;
    const int lane = tid & 31;
    const int half = lane >> 4;
    const int mn   = lane & 15;

    const int mblk  = blockIdx.x / 16;    // 288 super-tiles of 128 tokens
    const int ngrp  = blockIdx.x % 16;    // 16 groups of 64 features
    const int mbase = mblk * 128;
    const int ebase = ngrp * 64;

    auto issue = [&](int k0, int buf) {
        // A: 512 chunks (128 rows x 4 chunks of 8 halves); 2 per thread
#pragma unroll
        for (int q = 0; q < 2; ++q) {
            int c   = tid + q * 256;
            int row = c >> 2;
            int col = (c & 3) * 8;                      // halves
            async_ld_b128(&As[buf][row * KSTEP + col],
                          o + (size_t)(mbase + row) * DIMX + k0 + col);
        }
        // B: 256 chunks; 1 per thread
        {
            int row = tid >> 2;
            int col = (tid & 3) * 8;
            async_ld_b128(&Bs[buf][row * KSTEP + col],
                          wo + (size_t)(ebase + row) * DIMX + k0 + col);
        }
    };

    v8f acc[4] = {};
    int buf = 0;
    issue(0, 0);
    for (int k0 = 0; k0 < DIMX; k0 += KSTEP) {
        if (k0 + KSTEP < DIMX) {
            issue(k0 + KSTEP, buf ^ 1);
            asm volatile("s_wait_asynccnt 0x3" ::: "memory");
        } else {
            asm volatile("s_wait_asynccnt 0x0" ::: "memory");
        }
        __syncthreads();
        const _Float16* arow = &As[buf][(wv * 16 + mn) * KSTEP];
        v16h a = ld_a16(arow, 0, half);
#pragma unroll
        for (int j = 0; j < 4; ++j) {
            v16h b = ld_b16(&Bs[buf][(j * 16 + mn) * KSTEP], 0, half);
            acc[j] = wmma16(a, b, acc[j]);
        }
        __syncthreads();
        buf ^= 1;
    }
#pragma unroll
    for (int j = 0; j < 4; ++j)
#pragma unroll
        for (int r = 0; r < 8; ++r) {
            int tok = mbase + wv * 16 + r + half * 8;
            int e   = ebase + j * 16 + mn;
            out[(size_t)tok * DIMX + e] = acc[j][r] + bias[e];
        }
}

// ---------------- launcher ----------------
extern "C" void kernel_launch(void* const* d_in, const int* in_sizes, int n_in,
                              void* d_out, int out_size, void* d_ws, size_t ws_size,
                              hipStream_t stream) {
    const float* x     = (const float*)d_in[0];   // [2,32,18,32,1024]
    const float* w_qkv = (const float*)d_in[1];   // [3072,1024]
    const float* w_out = (const float*)d_in[2];   // [1024,1024]
    const float* b_out = (const float*)d_in[3];   // [1024]
    float* out = (float*)d_out;                   // [36864,1024] fp32

    // workspace partition (all offsets 256B-aligned), ~310 MB total
    char* ws = (char*)d_ws;
    const size_t WQ_BYTES  = (size_t)E3 * DIMX * 2;        //   6 MB
    const size_t WO_BYTES  = (size_t)DIMX * DIMX * 2;      //   2 MB
    const size_t QKV_BYTES = (size_t)NTOK * E3 * 2;        // 226 MB
    _Float16* wq_h  = (_Float16*)(ws);
    _Float16* wo_h  = (_Float16*)(ws + WQ_BYTES);
    _Float16* qkv_h = (_Float16*)(ws + WQ_BYTES + WO_BYTES);
    _Float16* o_h   = (_Float16*)(ws + WQ_BYTES + WO_BYTES + QKV_BYTES);

    // 0) weights -> f16
    {
        int n1 = E3 * DIMX;
        cvt_f16_kernel<<<(n1 + 255) / 256, 256, 0, stream>>>(w_qkv, wq_h, n1);
        int n2 = DIMX * DIMX;
        cvt_f16_kernel<<<(n2 + 255) / 256, 256, 0, stream>>>(w_out, wo_h, n2);
    }
    // 1) qkv projection: 288 x 48 blocks, 256 threads
    qkv_gemm_kernel<<<13824, 256, 0, stream>>>(x, wq_h, qkv_h);
    // 2) attention: 18432 waves, 4 waves/block
    attn_kernel<<<4608, 128, 0, stream>>>(qkv_h, o_h);
    // 3) output projection: 288 x 16 blocks, 256 threads
    out_gemm_kernel<<<4608, 256, 0, stream>>>(o_h, wo_h, b_out, out);
}
